// MultiheadAttention_83021717832009
// MI455X (gfx1250) — compile-verified, matrix operand order checked by reading, not run on hardware
//
#include <hip/hip_runtime.h>

// MI455X (gfx1250) fused multihead-attention forward.
// fp32 ref -> f16 operands + f32 WMMA accumulation (v_wmma_f32_16x16x32_f16).
// Flash-attention fusion avoids 2x537MB of score traffic on HBM.
// K/V tiles staged to LDS by the Tensor Data Mover (TENSORcnt), falling back
// to per-lane async global->LDS copies (ASYNCcnt). GEMMs ping-pong pipelined.

#define Tq 2048
#define Ss 2048
#define Bb 2
#define Ee 1024
#define Hh 16
#define Dd 64

#define USE_TDM 1

typedef __attribute__((ext_vector_type(16))) _Float16     v16h;
typedef __attribute__((ext_vector_type(8)))  _Float16     h8;
typedef __attribute__((ext_vector_type(8)))  float        v8f;
typedef __attribute__((ext_vector_type(4)))  int          i32x4;
typedef __attribute__((ext_vector_type(8)))  int          i32x8;
typedef __attribute__((ext_vector_type(4)))  unsigned int u32x4;
typedef __attribute__((address_space(1)))    i32x4        gl_i32x4;
typedef __attribute__((address_space(3)))    i32x4        ls_i32x4;

#if defined(__has_builtin)
#if __has_builtin(__builtin_amdgcn_global_load_async_to_lds_b128)
#define HAVE_ASYNC_LDS 1
#endif
#if __has_builtin(__builtin_amdgcn_s_wait_asynccnt)
#define HAVE_WAIT_ASYNC 1
#endif
#if USE_TDM && __has_builtin(__builtin_amdgcn_tensor_load_to_lds)
#define HAVE_TDM 1
#endif
#endif

// ---------------------------------------------------------------------------
// Async global->LDS 16-byte copy (per lane); ASYNCcnt-tracked. Fallback path.
// ---------------------------------------------------------------------------
__device__ __forceinline__ void async_cp16(const _Float16* g, _Float16* l) {
#ifdef HAVE_ASYNC_LDS
    __builtin_amdgcn_global_load_async_to_lds_b128(
        (gl_i32x4*)(uintptr_t)g,
        (ls_i32x4*)(uint32_t)(uintptr_t)l,
        /*offset=*/0, /*cpol=*/0);
#else
    *(h8*)l = *(const h8*)g;
#endif
}

#ifdef HAVE_WAIT_ASYNC
#define WAIT_ASYNC(n) __builtin_amdgcn_s_wait_asynccnt(n)
#else
#define WAIT_ASYNC(n) asm volatile("s_wait_asynccnt " #n ::: "memory")
#endif

// ---------------------------------------------------------------------------
// TDM: one descriptor loads a 64(row) x 64(elem) f16 tile into LDS with
// 16B row padding (128B data + 16B pad = 144B = 72-half rows).
// D# packing per CDNA5 ISA 8.3/8.4: count=1, type=2, data_size=1 (2B),
// pad_interval=4 (32 dwords), pad_amount=3 (4 dwords).
// ---------------------------------------------------------------------------
#ifdef HAVE_TDM
__device__ __forceinline__ void tdm_load_tile64(
    const _Float16* g, uint32_t lds_off,
    uint32_t tensor_d0, uint32_t row_stride, uint32_t tensor_rows)
{
    const uint64_t ga = (uint64_t)(uintptr_t)g;
    u32x4 g0;
    g0[0] = 1u;                                   // count=1 (valid descriptor)
    g0[1] = lds_off;                              // lds_addr (bytes)
    g0[2] = (uint32_t)ga;                         // global_addr[31:0]
    g0[3] = (uint32_t)(ga >> 32) | (2u << 30);    // global_addr[56:32] | type=2

    i32x8 g1;
    g1[0] = (int)((1u << 16) | (1u << 20) | (4u << 22) | (3u << 25));
    g1[1] = (int)((tensor_d0 & 0xFFFFu) << 16);   // tensor_dim0[15:0]
    g1[2] = (int)((tensor_d0 >> 16) | ((tensor_rows & 0xFFFFu) << 16));
    g1[3] = (int)((tensor_rows >> 16) | (64u << 16));  // tile_dim0 = 64
    g1[4] = (int)64;                              // tile_dim1 = 64, tile_dim2 = 0
    g1[5] = (int)row_stride;                      // tensor_dim0_stride[31:0]
    g1[6] = 0;                                    // stride[47:32] | dim1_stride lo
    g1[7] = 0;

    i32x4 zz = {};
#if __clang_major__ >= 23
    i32x8 z8 = {};
    __builtin_amdgcn_tensor_load_to_lds(g0, g1, zz, zz, z8, 0);
#else
    __builtin_amdgcn_tensor_load_to_lds(g0, g1, zz, zz, 0);
#endif
}
#endif

// ---------------------------------------------------------------------------
// WMMA helper: D = A(16x32 f16) x B(32x16 f16) + C(16x16 f32)
// ---------------------------------------------------------------------------
__device__ __forceinline__ v8f wmma_f16(v16h a, v16h b, v8f c) {
    return __builtin_amdgcn_wmma_f32_16x16x32_f16(
        /*neg_a=*/false, a, /*neg_b=*/false, b,
        /*c_mod=*/(short)0, c, /*reuse_a=*/false, /*reuse_b=*/false);
}

// ---------------------------------------------------------------------------
// Fragment loader: lane l holds row (row16 + (l&15)); halves live at
// k = kbase + 8*(l>>4)..+7 and k = kbase + 16 + 8*(l>>4)..+7
// -> two contiguous 16B chunks (2x b128). Global or LDS pointers.
// ---------------------------------------------------------------------------
__device__ __forceinline__ v16h ld_frag(const _Float16* base, int row16,
                                        int rstride, int kbase) {
    const int lane = threadIdx.x & 31;
    const _Float16* p = base + (size_t)(row16 + (lane & 15)) * rstride
                             + kbase + ((lane >> 4) << 3);
    h8 lo = *(const h8*)(p);
    h8 hi = *(const h8*)(p + 16);
    return __builtin_shufflevector(lo, hi, 0,1,2,3,4,5,6,7,
                                            8,9,10,11,12,13,14,15);
}

// ---------------------------------------------------------------------------
// Kernel 0: fp32 -> f16 conversion (inputs and weights)
// ---------------------------------------------------------------------------
__global__ void cvt_f32_f16(const float* __restrict__ src,
                            _Float16* __restrict__ dst, int n) {
    int i = blockIdx.x * blockDim.x + threadIdx.x;
    if (i < n) dst[i] = (_Float16)src[i];
}

// ---------------------------------------------------------------------------
// Kernel 1: fused QKV projection. z = 0/1/2 -> Q/K/V.
// Batch fixed per block; V-transpose epilogue packs 8 consecutive t per lane
// into one b128 store. Ping-pong double-buffered k-loop.
// ---------------------------------------------------------------------------
__global__ __launch_bounds__(256) void qkv_proj_kernel(
    const _Float16* __restrict__ Xq, const _Float16* __restrict__ Xk,
    const _Float16* __restrict__ Xv,
    const _Float16* __restrict__ Wq, const _Float16* __restrict__ Wk,
    const _Float16* __restrict__ Wv,
    const float* __restrict__ bq, const float* __restrict__ bk,
    const float* __restrict__ bv,
    _Float16* __restrict__ Qh, _Float16* __restrict__ Kh,
    _Float16* __restrict__ Vt)
{
    const int z = blockIdx.z;
    const _Float16* X = (z == 0) ? Xq : (z == 1) ? Xk : Xv;
    const _Float16* W = (z == 0) ? Wq : (z == 1) ? Wk : Wv;
    const float* bias = (z == 0) ? bq : (z == 1) ? bk : bv;

    const int wave = threadIdx.x >> 5;
    const int lane = threadIdx.x & 31;
    const int g  = lane >> 4;
    const int nl = lane & 15;
    const int b2    = blockIdx.y & 1;
    const int trow  = (blockIdx.y >> 1) * 128 + wave * 16;
    const int nbase = blockIdx.x * 64;
    const _Float16* Xb = X + (size_t)b2 * Ee;

    v8f acc[4] = {};
    v16h a0 = ld_frag(Xb, trow, Bb * Ee, 0);
    v16h b0[4];
#pragma unroll
    for (int j = 0; j < 4; ++j) b0[j] = ld_frag(W, nbase + j * 16, Ee, 0);

    for (int kb = 0; kb < Ee; kb += 64) {
        v16h a1 = ld_frag(Xb, trow, Bb * Ee, kb + 32);
        v16h b1[4];
#pragma unroll
        for (int j = 0; j < 4; ++j)
            b1[j] = ld_frag(W, nbase + j * 16, Ee, kb + 32);
        __builtin_prefetch(W + (size_t)(nbase + nl) * Ee + kb + 64, 0, 1);
#pragma unroll
        for (int j = 0; j < 4; ++j) acc[j] = wmma_f16(a0, b0[j], acc[j]);
        const int kn = kb + 64;
        if (kn < Ee) {
            a0 = ld_frag(Xb, trow, Bb * Ee, kn);
#pragma unroll
            for (int j = 0; j < 4; ++j)
                b0[j] = ld_frag(W, nbase + j * 16, Ee, kn);
        }
#pragma unroll
        for (int j = 0; j < 4; ++j) acc[j] = wmma_f16(a1, b1[j], acc[j]);
    }

    const float scale = (z == 0) ? 0.125f : 1.0f;      // 1/sqrt(64)
#pragma unroll
    for (int j = 0; j < 4; ++j) {
        const int gn = nbase + j * 16 + nl;
        const float bias_n = bias[gn];
        const int h  = gn >> 6;
        const int id = gn & 63;
        const size_t bh = (size_t)(b2 * Hh + h);
        if (z == 2) {
            h8 vv;
#pragma unroll
            for (int r = 0; r < 8; ++r) vv[r] = (_Float16)(acc[j][r] + bias_n);
            *(h8*)(Vt + (bh * Dd + id) * Ss + trow + g * 8) = vv;
        } else {
#pragma unroll
            for (int r = 0; r < 8; ++r) {
                const int t = trow + g * 8 + r;
                const _Float16 hv = (_Float16)((acc[j][r] + bias_n) * scale);
                if (z == 0) Qh[(bh * Tq + t) * Dd + id] = hv;
                else        Kh[(bh * Ss + t) * Dd + id] = hv;
            }
        }
    }
}

// ---------------------------------------------------------------------------
// Kernel 2: flash attention. Block = one (b,h) x 128 query rows; 8 waves,
// each wave owns 16 query rows. S-tiles of 64, double-buffered in LDS:
// TDM descriptors (wave 0, TENSORcnt) or per-lane async copies (ASYNCcnt).
// ---------------------------------------------------------------------------
__global__ __launch_bounds__(256) void flash_attn_kernel(
    const _Float16* __restrict__ Qh, const _Float16* __restrict__ Kh,
    const _Float16* __restrict__ Vt,
    const float* __restrict__ amask, const unsigned char* __restrict__ kpm,
    _Float16* __restrict__ Oa)
{
    __shared__ _Float16 sK[2][64][72];   // K tile: rows = s_local, cols = d
    __shared__ _Float16 sV[2][64][72];   // V tile: rows = d,      cols = s_local
    __shared__ _Float16 sP[8][16][72];   // per-wave P tile (C->A relayout)

    const int tid  = threadIdx.x;
    const int wave = tid >> 5;
    const int lane = tid & 31;
    const int g  = lane >> 4;
    const int nl = lane & 15;
    const int bh = blockIdx.y;
    const int b  = bh / Hh;
    const int h  = bh % Hh;
    const int trow = blockIdx.x * 128 + wave * 16;

    const _Float16* Qb = Qh + (size_t)bh * Tq * Dd;
    const _Float16* Kb = Kh + (size_t)bh * Ss * Dd;
    const _Float16* Vb = Vt + (size_t)bh * Dd * Ss;

    const v16h aq0 = ld_frag(Qb, trow, Dd, 0);
    const v16h aq1 = ld_frag(Qb, trow, Dd, 32);

    v8f O[4] = {};
    float m_run[8], l_run[8];
#pragma unroll
    for (int r = 0; r < 8; ++r) { m_run[r] = -3.0e38f; l_run[r] = 0.0f; }

#ifdef HAVE_TDM
    // wave 0 issues 2 TDM descriptors per tile; TDM applies the row padding
    auto stage = [&](int buf, int s0) {
        if (wave == 0) {
            tdm_load_tile64(Kb + (size_t)s0 * Dd,
                            (uint32_t)(uintptr_t)&sK[buf][0][0],
                            /*tensor_d0=*/Dd, /*row_stride=*/Dd,
                            /*tensor_rows=*/Ss - s0);
            tdm_load_tile64(Vb + s0,
                            (uint32_t)(uintptr_t)&sV[buf][0][0],
                            /*tensor_d0=*/Ss - s0, /*row_stride=*/Ss,
                            /*tensor_rows=*/Dd);
        }
    };
#else
    auto stage = [&](int buf, int s0) {
#pragma unroll
        for (int q = 0; q < 2; ++q) {
            const int c = tid + q * 256;
            const int row = c >> 3;
            const int col8 = (c & 7) * 8;
            async_cp16(Kb + (size_t)(s0 + row) * Dd + col8,
                       &sK[buf][row][col8]);
        }
#pragma unroll
        for (int q = 0; q < 2; ++q) {
            const int c = tid + q * 256;
            const int row = c >> 3;
            const int col8 = (c & 7) * 8;
            async_cp16(Vb + (size_t)row * Ss + s0 + col8,
                       &sV[buf][row][col8]);
        }
    };
#endif

    const int ntiles = Ss / 64;
    stage(0, 0);

    for (int it = 0; it < ntiles; ++it) {
        const int s0  = it * 64;
        const int cur = it & 1;
#ifdef HAVE_TDM
        if (it + 1 < ntiles) {
            stage(cur ^ 1, s0 + 64);
            if (wave == 0) __builtin_amdgcn_s_wait_tensorcnt(2);
        } else {
            if (wave == 0) __builtin_amdgcn_s_wait_tensorcnt(0);
        }
#else
        if (it + 1 < ntiles) {
            stage(cur ^ 1, s0 + 64);
            WAIT_ASYNC(4);
        } else {
            WAIT_ASYNC(0);
        }
#endif
        __syncthreads();               // staged tile visible to all waves

        // ---- scores: 16x64 = 4 N-frags, K over d=64 in 2 chunks ----
        v8f sc[4] = {};
#pragma unroll
        for (int j = 0; j < 4; ++j) {
            v16h bk0 = ld_frag(&sK[cur][0][0], j * 16, 72, 0);
            sc[j] = wmma_f16(aq0, bk0, sc[j]);
            v16h bk1 = ld_frag(&sK[cur][0][0], j * 16, 72, 32);
            sc[j] = wmma_f16(aq1, bk1, sc[j]);
        }
        // ---- masks: additive attn_mask + key padding penalty ----
#pragma unroll
        for (int j = 0; j < 4; ++j) {
            const int s = s0 + j * 16 + nl;
            const float kpen = kpm[b * Ss + s] ? -1.0e9f : 0.0f;
#pragma unroll
            for (int r = 0; r < 8; ++r) {
                const int t = trow + g * 8 + r;
                sc[j][r] += amask[(size_t)t * Ss + s] + kpen;
            }
        }
        // ---- online softmax (row = fixed (g,r); 16 lanes span N) ----
#pragma unroll
        for (int r = 0; r < 8; ++r) {
            float vmax = fmaxf(fmaxf(sc[0][r], sc[1][r]),
                               fmaxf(sc[2][r], sc[3][r]));
#pragma unroll
            for (int xm = 1; xm < 16; xm <<= 1)
                vmax = fmaxf(vmax, __shfl_xor(vmax, xm, 32));
            const float mn = fmaxf(m_run[r], vmax);
            const float alpha = __expf(m_run[r] - mn);
            float rs = 0.0f;
#pragma unroll
            for (int j = 0; j < 4; ++j) {
                const float p = __expf(sc[j][r] - mn);
                sc[j][r] = p;
                rs += p;
            }
#pragma unroll
            for (int xm = 1; xm < 16; xm <<= 1)
                rs += __shfl_xor(rs, xm, 32);
            l_run[r] = l_run[r] * alpha + rs;
            m_run[r] = mn;
#pragma unroll
            for (int j = 0; j < 4; ++j) O[j][r] *= alpha;
        }
        // ---- P: C-layout -> LDS -> A-layout (same wave, no barrier) ----
#pragma unroll
        for (int j = 0; j < 4; ++j)
#pragma unroll
            for (int r = 0; r < 8; ++r)
                sP[wave][g * 8 + r][j * 16 + nl] = (_Float16)sc[j][r];
        asm volatile("s_wait_dscnt 0x0" ::: "memory");
        const v16h ap0 = ld_frag(&sP[wave][0][0], 0, 72, 0);
        const v16h ap1 = ld_frag(&sP[wave][0][0], 0, 72, 32);
        // ---- O += P @ V ----
#pragma unroll
        for (int j = 0; j < 4; ++j) {
            v16h bv0 = ld_frag(&sV[cur][0][0], j * 16, 72, 0);
            O[j] = wmma_f16(ap0, bv0, O[j]);
            v16h bv1 = ld_frag(&sV[cur][0][0], j * 16, 72, 32);
            O[j] = wmma_f16(ap1, bv1, O[j]);
        }
        __syncthreads();   // tile fully consumed before its buffer is reused
    }

    // ---- normalize and store attention output as f16 [T,B,E] ----
#pragma unroll
    for (int r = 0; r < 8; ++r) {
        const float inv = 1.0f / l_run[r];
        const int t = trow + g * 8 + r;
#pragma unroll
        for (int j = 0; j < 4; ++j) {
            const int e = h * Dd + j * 16 + nl;
            Oa[((size_t)t * Bb + b) * Ee + e] = (_Float16)(O[j][r] * inv);
        }
    }
}

// ---------------------------------------------------------------------------
// Kernel 3: output projection  out = Oa @ Wo^T + bo   (fp32 result)
// ---------------------------------------------------------------------------
__global__ __launch_bounds__(256) void out_proj_kernel(
    const _Float16* __restrict__ Oa, const _Float16* __restrict__ Wo,
    const float* __restrict__ bo, float* __restrict__ out)
{
    const int wave = threadIdx.x >> 5;
    const int lane = threadIdx.x & 31;
    const int g  = lane >> 4;
    const int nl = lane & 15;
    const int b2    = blockIdx.y & 1;
    const int trow  = (blockIdx.y >> 1) * 128 + wave * 16;
    const int nbase = blockIdx.x * 64;
    const _Float16* Ob = Oa + (size_t)b2 * Ee;

    v8f acc[4] = {};
    v16h a0 = ld_frag(Ob, trow, Bb * Ee, 0);
    v16h b0[4];
#pragma unroll
    for (int j = 0; j < 4; ++j) b0[j] = ld_frag(Wo, nbase + j * 16, Ee, 0);

    for (int kb = 0; kb < Ee; kb += 64) {
        v16h a1 = ld_frag(Ob, trow, Bb * Ee, kb + 32);
        v16h b1[4];
#pragma unroll
        for (int j = 0; j < 4; ++j)
            b1[j] = ld_frag(Wo, nbase + j * 16, Ee, kb + 32);
        __builtin_prefetch(Wo + (size_t)(nbase + nl) * Ee + kb + 64, 0, 1);
#pragma unroll
        for (int j = 0; j < 4; ++j) acc[j] = wmma_f16(a0, b0[j], acc[j]);
        const int kn = kb + 64;
        if (kn < Ee) {
            a0 = ld_frag(Ob, trow, Bb * Ee, kn);
#pragma unroll
            for (int j = 0; j < 4; ++j)
                b0[j] = ld_frag(Wo, nbase + j * 16, Ee, kn);
        }
#pragma unroll
        for (int j = 0; j < 4; ++j) acc[j] = wmma_f16(a1, b1[j], acc[j]);
    }

#pragma unroll
    for (int j = 0; j < 4; ++j) {
        const int gn = nbase + j * 16 + nl;
        const float bias_n = bo[gn];
#pragma unroll
        for (int r = 0; r < 8; ++r) {
            const int t = trow + g * 8 + r;
            out[((size_t)t * Bb + b2) * Ee + gn] = acc[j][r] + bias_n;
        }
    }
}

// ---------------------------------------------------------------------------
// Host launcher. Workspace (f16): xq/xk/xv 3x4M, weights 4x1M,
// Qh/Kh/Vt 3x4M, Oa 4M  -> 32M f16 = 64 MB.
// ---------------------------------------------------------------------------
extern "C" void kernel_launch(void* const* d_in, const int* in_sizes, int n_in,
                              void* d_out, int out_size, void* d_ws,
                              size_t ws_size, hipStream_t stream) {
    const float* query = (const float*)d_in[0];
    const float* key_  = (const float*)d_in[1];
    const float* value = (const float*)d_in[2];
    const float* amask = (const float*)d_in[3];
    const unsigned char* kpm = (const unsigned char*)d_in[4];
    const float* Wq = (const float*)d_in[5];
    const float* bq = (const float*)d_in[6];
    const float* Wk = (const float*)d_in[7];
    const float* bk = (const float*)d_in[8];
    const float* Wv = (const float*)d_in[9];
    const float* bv = (const float*)d_in[10];
    const float* Wo = (const float*)d_in[11];
    const float* bo = (const float*)d_in[12];
    float* out = (float*)d_out;

    const size_t TBE = (size_t)Tq * Bb * Ee;   // 4M
    const size_t EE  = (size_t)Ee * Ee;        // 1M
    _Float16* ws  = (_Float16*)d_ws;
    _Float16* xq  = ws;
    _Float16* xk  = xq  + TBE;
    _Float16* xv  = xk  + TBE;
    _Float16* wqh = xv  + TBE;
    _Float16* wkh = wqh + EE;
    _Float16* wvh = wkh + EE;
    _Float16* woh = wvh + EE;
    _Float16* qh  = woh + EE;
    _Float16* kh  = qh  + TBE;
    _Float16* vt  = kh  + TBE;
    _Float16* oa  = vt  + TBE;

    const int cb = 256;
    const int nTBE = (int)TBE, nEE = (int)EE;
    cvt_f32_f16<<<(nTBE + cb - 1) / cb, cb, 0, stream>>>(query, xq, nTBE);
    cvt_f32_f16<<<(nTBE + cb - 1) / cb, cb, 0, stream>>>(key_,  xk, nTBE);
    cvt_f32_f16<<<(nTBE + cb - 1) / cb, cb, 0, stream>>>(value, xv, nTBE);
    cvt_f32_f16<<<(nEE  + cb - 1) / cb, cb, 0, stream>>>(Wq, wqh, nEE);
    cvt_f32_f16<<<(nEE  + cb - 1) / cb, cb, 0, stream>>>(Wk, wkh, nEE);
    cvt_f32_f16<<<(nEE  + cb - 1) / cb, cb, 0, stream>>>(Wv, wvh, nEE);
    cvt_f32_f16<<<(nEE  + cb - 1) / cb, cb, 0, stream>>>(Wo, woh, nEE);

    // QKV projections: grid (N/64, (T/128)*B, 3)
    qkv_proj_kernel<<<dim3(Ee / 64, (Tq / 128) * Bb, 3), 256, 0, stream>>>(
        xq, xk, xv, wqh, wkh, wvh, bq, bk, bv, qh, kh, vt);

    // Flash attention: grid (T/128, B*H)
    flash_attn_kernel<<<dim3(Tq / 128, Bb * Hh), 256, 0, stream>>>(
        qh, kh, vt, amask, kpm, oa);

    // Output projection: grid (N/64, (T/128)*B)
    out_proj_kernel<<<dim3(Ee / 64, (Tq / 128) * Bb), 256, 0, stream>>>(
        oa, woh, bo, out);
}
